// GCN_16501264351451
// MI455X (gfx1250) — compile-verified, matrix operand order checked by reading, not run on hardware
//
#include <hip/hip_runtime.h>

typedef __attribute__((ext_vector_type(16))) _Float16 v16h;
typedef __attribute__((ext_vector_type(8)))  _Float16 v8h;
typedef __attribute__((ext_vector_type(8)))  float    v8f;

#define N_NODES 50000
#define N_GRAPHS 500
#define N_EDGES 800000
#define DIM 128

// ---------------------------------------------------------------------------
// Degree accumulation: out_deg[src[e]] += 1 ; in_deg[dst[e]] += 1
// ---------------------------------------------------------------------------
__global__ void gcn_deg_kernel(const int* __restrict__ src,
                               const int* __restrict__ dst,
                               float* __restrict__ out_deg,
                               float* __restrict__ in_deg) {
    int e = blockIdx.x * blockDim.x + threadIdx.x;
    if (e < N_EDGES) {
        atomicAdd(&out_deg[src[e]], 1.0f);
        atomicAdd(&in_deg[dst[e]], 1.0f);
    }
}

// deg -> rsqrt(max(deg,1)) in place over 2*N contiguous floats (out_inv|in_inv)
__global__ void gcn_inv_kernel(float* __restrict__ d) {
    int i = blockIdx.x * blockDim.x + threadIdx.x;
    if (i < 2 * N_NODES) {
        d[i] = rsqrtf(fmaxf(d[i], 1.0f));
    }
}

// ---------------------------------------------------------------------------
// Pre-pack weights: W (f32, [K][N] row-major) -> Wt (f16, [N][K]).
// 64KB -> 32KB, done once per layer; makes every B-fragment load in the
// GEMM two contiguous global_load_b128s instead of 16 strided b32s.
// ---------------------------------------------------------------------------
__global__ void gcn_pack_w_kernel(const float* __restrict__ W,
                                  _Float16* __restrict__ Wt) {
    int i = blockIdx.x * blockDim.x + threadIdx.x;   // i = n*128 + k
    if (i < DIM * DIM) {
        int n = i >> 7;
        int k = i & (DIM - 1);
        Wt[i] = (_Float16)W[(size_t)k * DIM + n];
    }
}

// ---------------------------------------------------------------------------
// Message passing: agg[dst[e]][d] += x[src[e]][d] * out_inv[src[e]]
// float4 granularity: one b128 load + 4 global_atomic_add_f32 per thread.
// Atomics land in L2 (agg = 25.6MB << 192MB L2).
// ---------------------------------------------------------------------------
__global__ void gcn_scatter_kernel(const float* __restrict__ x,
                                   const int* __restrict__ src,
                                   const int* __restrict__ dst,
                                   const float* __restrict__ out_inv,
                                   float* __restrict__ agg) {
    long long tid = (long long)blockIdx.x * blockDim.x + threadIdx.x;
    if (tid >= (long long)N_EDGES * (DIM / 4)) return;
    int e = (int)(tid >> 5);              // DIM/4 == 32 quads per edge
    int q = (int)(tid & 31) << 2;         // starting feature index
    int s = src[e];
    float sc = out_inv[s];
    float4 v = *(const float4*)(x + (size_t)s * DIM + q);
    float* ap = agg + (size_t)dst[e] * DIM + q;
    atomicAdd(ap + 0, v.x * sc);
    atomicAdd(ap + 1, v.y * sc);
    atomicAdd(ap + 2, v.z * sc);
    atomicAdd(ap + 3, v.w * sc);
}

// ---------------------------------------------------------------------------
// Fused: h = PReLU( (agg * in_inv[:,None]) @ W )  via v_wmma_f32_16x16x32_f16
// One wave per 16x16 output tile; K=128 -> 4 WMMA steps, f32 accumulate.
// A layout (16-bit, 16x32): lanes 0-15 rows M, halves = K {0..7,16..23};
// lanes 16-31 same rows, K {8..15,24..31}. B mirrored with N per lane,
// served from the pre-packed f16 [N][K] buffer (contiguous 16B loads).
// C/D: vgpr r, lanes 0-15 -> (M=r, N=lane); lanes 16-31 -> (M=r+8, N=lane-16).
// ---------------------------------------------------------------------------
__global__ __launch_bounds__(32) void gcn_gemm_prelu_wmma(
    const float* __restrict__ agg, const float* __restrict__ in_inv,
    const _Float16* __restrict__ Wt, const float* __restrict__ a_slope,
    float* __restrict__ out) {
    const int lane = threadIdx.x & 31;
    const int half = lane >> 4;      // 0: K-low group, 1: K-high group
    const int idx  = lane & 15;
    const int m0 = blockIdx.x * 16;
    const int n0 = blockIdx.y * 16;

    const int m = m0 + idx;          // A row owned by this lane
    const int n = n0 + idx;          // B col owned by this lane
    const float scale = in_inv[m];
    const float* __restrict__ arow = agg + (size_t)m * DIM;
    const _Float16* __restrict__ wrow = Wt + (size_t)n * DIM;

    v8f c = {};
#pragma unroll
    for (int kb = 0; kb < 4; ++kb) {
        const int k0 = kb * 32 + half * 8;

        // A: 16 f32 in two 32B chunks -> scale -> f16 fragment
        float4 a0 = *(const float4*)(arow + k0);
        float4 a1 = *(const float4*)(arow + k0 + 4);
        float4 a2 = *(const float4*)(arow + k0 + 16);
        float4 a3 = *(const float4*)(arow + k0 + 20);
        v16h a;
        a[0]  = (_Float16)(a0.x * scale); a[1]  = (_Float16)(a0.y * scale);
        a[2]  = (_Float16)(a0.z * scale); a[3]  = (_Float16)(a0.w * scale);
        a[4]  = (_Float16)(a1.x * scale); a[5]  = (_Float16)(a1.y * scale);
        a[6]  = (_Float16)(a1.z * scale); a[7]  = (_Float16)(a1.w * scale);
        a[8]  = (_Float16)(a2.x * scale); a[9]  = (_Float16)(a2.y * scale);
        a[10] = (_Float16)(a2.z * scale); a[11] = (_Float16)(a2.w * scale);
        a[12] = (_Float16)(a3.x * scale); a[13] = (_Float16)(a3.y * scale);
        a[14] = (_Float16)(a3.z * scale); a[15] = (_Float16)(a3.w * scale);

        // B: two contiguous 16B f16 loads
        v8h blo = *(const v8h*)(wrow + k0);
        v8h bhi = *(const v8h*)(wrow + k0 + 16);
        v16h b;
#pragma unroll
        for (int j = 0; j < 8; ++j) { b[j] = blo[j]; b[8 + j] = bhi[j]; }

        c = __builtin_amdgcn_wmma_f32_16x16x32_f16(
            /*neg_a=*/false, a, /*neg_b=*/false, b,
            /*c_mod=*/(short)0, c, /*reuse_a=*/false, /*reuse_b=*/false);
    }

    const float slope = a_slope[0];
    const int mb = m0 + half * 8;
#pragma unroll
    for (int r = 0; r < 8; ++r) {
        float z = c[r];
        out[(size_t)(mb + r) * DIM + (n0 + idx)] = (z > 0.0f) ? z : slope * z;
    }
}

// ---------------------------------------------------------------------------
// Sum pooling: hg[graph_id[n]][col0 + d] += h[n][d]   (hg row stride 384)
// float4 granularity.
// ---------------------------------------------------------------------------
__global__ void gcn_pool_kernel(const float* __restrict__ h,
                                const int* __restrict__ gid,
                                float* __restrict__ hg, int col0) {
    long long tid = (long long)blockIdx.x * blockDim.x + threadIdx.x;
    if (tid >= (long long)N_NODES * (DIM / 4)) return;
    int n = (int)(tid >> 5);
    int q = (int)(tid & 31) << 2;
    float4 v = *(const float4*)(h + (size_t)n * DIM + q);
    float* gp = hg + (size_t)gid[n] * (3 * DIM) + col0 + q;
    atomicAdd(gp + 0, v.x);
    atomicAdd(gp + 1, v.y);
    atomicAdd(gp + 2, v.z);
    atomicAdd(gp + 3, v.w);
}

// ---------------------------------------------------------------------------
extern "C" void kernel_launch(void* const* d_in, const int* in_sizes, int n_in,
                              void* d_out, int out_size, void* d_ws, size_t ws_size,
                              hipStream_t stream) {
    const float* feat = (const float*)d_in[0];
    const float* Wm[3] = {(const float*)d_in[1], (const float*)d_in[2], (const float*)d_in[3]};
    const float* am[3] = {(const float*)d_in[4], (const float*)d_in[5], (const float*)d_in[6]};
    const int* src = (const int*)d_in[7];
    const int* dst = (const int*)d_in[8];
    const int* gid = (const int*)d_in[9];

    float* ws = (float*)d_ws;
    float* out_inv = ws;                                   // [N]
    float* in_inv  = ws + N_NODES;                         // [N]
    float* agg     = ws + 2 * N_NODES;                     // [N, D]
    float* hbuf    = agg + (size_t)N_NODES * DIM;          // [N, D]
    _Float16* wt   = (_Float16*)(hbuf + (size_t)N_NODES * DIM);  // [D, D] f16

    float* h_out = (float*)d_out;                          // [N, D] final layer
    float* hg    = (float*)d_out + (size_t)N_NODES * DIM;  // [G, 3D]

    // Degrees -> normalizers
    hipMemsetAsync(out_inv, 0, 2 * N_NODES * sizeof(float), stream);
    gcn_deg_kernel<<<(N_EDGES + 255) / 256, 256, 0, stream>>>(src, dst, out_inv, in_inv);
    gcn_inv_kernel<<<(2 * N_NODES + 255) / 256, 256, 0, stream>>>(out_inv);

    // Zero pooled output (harness does not re-zero between replays)
    hipMemsetAsync(hg, 0, (size_t)N_GRAPHS * 3 * DIM * sizeof(float), stream);

    const int scatter_blocks = (int)(((long long)N_EDGES * (DIM / 4) + 255) / 256);
    const int pool_blocks = (int)(((long long)N_NODES * (DIM / 4) + 255) / 256);
    const int packw_blocks = (DIM * DIM + 255) / 256;
    dim3 gemm_grid(N_NODES / 16, DIM / 16);

    const float* x = feat;
    for (int l = 0; l < 3; ++l) {
        float* h = (l == 2) ? h_out : hbuf;
        gcn_pack_w_kernel<<<packw_blocks, 256, 0, stream>>>(Wm[l], wt);
        hipMemsetAsync(agg, 0, (size_t)N_NODES * DIM * sizeof(float), stream);
        gcn_scatter_kernel<<<scatter_blocks, 256, 0, stream>>>(x, src, dst, out_inv, agg);
        gcn_gemm_prelu_wmma<<<gemm_grid, 32, 0, stream>>>(agg, in_inv, wt, am[l], h);
        gcn_pool_kernel<<<pool_blocks, 256, 0, stream>>>(h, gid, hg, l * DIM);
        x = h;  // layer l+1 consumes h (hbuf safely overwritten after scatter)
    }
}
